// MutualRefineAndPooling_47124381172076
// MI455X (gfx1250) — matrix operand order repaired and checked by reading, not compiled, not run on hardware
//
#include <hip/hip_runtime.h>
#include <cstdint>
#include <cstddef>

// ---------------------------------------------------------------------------
// MutualRefineAndPooling fused kernel for gfx1250 (MI455X), bf16 WMMA.
//
// v2: 32 rows per workgroup, two 16-row strips share every B-fragment in
// registers (dual-accumulator WMMA) -> halves L2 weight traffic, which was
// the limiting term (~14 GB L2 reads vs ~1.1 GB HBM at 23.3 TB/s).
// LDS ~118 KB via dynamic shared memory (gfx1250 allows 320 KB/workgroup;
// 2 workgroups/WGP -> 16 waves/WGP for latency hiding).
//
// Math simplification (exact): pooling attention has nhead=1, so
//   pooled = (sum_m softmax_m(q.k_m) * kv_m) @ Wv_p^T + bv_p
// with s_m = (q @ Wk_p) . kv_m + const  (const cancels in softmax).
// ---------------------------------------------------------------------------

#define DMODEL 256
#define TT 3
#define MM 5
#define BB 32768
#define ROWS 32   // rows per workgroup (two 16-row WMMA strips)
#define SAS 1288  // LDS activation row stride (bf16 elems); mult of 8, bank pad

typedef __bf16 bf16_t;
typedef bf16_t v16bf __attribute__((ext_vector_type(16)));
typedef float v8f __attribute__((ext_vector_type(8)));

__device__ __forceinline__ unsigned short f32_to_bf16(float f) {
  unsigned int u = __float_as_uint(f);
  unsigned int rounded = u + 0x7FFFu + ((u >> 16) & 1u); // RNE
  bool is_nan = (u & 0x7FFFFFFFu) > 0x7F800000u;
  unsigned int r = is_nan ? (u | 0x00400000u) : rounded;
  return (unsigned short)(r >> 16);
}
__device__ __forceinline__ float bf16_to_f32(unsigned short h) {
  return __uint_as_float(((unsigned int)h) << 16);
}

struct Params {
  float* out;
  const float* src_walk; const float* dst_walk;
  const float* src_pt;   const float* dst_pt;
  const unsigned char* src_masks; const unsigned char* dst_masks;
  const unsigned short* ws; // bf16 weights
  const float* qk;          // precomputed (q @ Wk_p) / 16
  const float* bv_c[2]; const float* bo_c[2]; const float* bg[2];
  const float* bv_p;    const float* bo_p;
  const float* bf1[2];  const float* lng1[2]; const float* lnb1[2]; const float* bf2[2];
  const float* bffn1[2]; const float* bffn2[2];
  const float* lng[2];  const float* lnb[2];
  size_t wv_c[2], wo_c[2], wg[2], wf1[2], wf2[2], wffn1[2], wffn2[2], wv_p, wo_p;
};

// Dual-strip WMMA: two 16-row A strips (strip1 = strip0 + 16 LDS rows) share
// each B fragment.  A layout per CDNA5 16-bit A 16x32; B: lanes 0-15 hold
// K 0..7,16..23 of column n, lanes 16-31 hold K 8..15,24..31.
__device__ __forceinline__ void wmma_tile2(const unsigned short* __restrict__ sA,
                                           int aCol, int K,
                                           const unsigned short* __restrict__ W,
                                           int ldk, int nOff, int lane,
                                           v8f& acc0, v8f& acc1) {
  const int row = lane & 15;
  const int kbA = (lane >> 4) << 3;
  const int n   = nOff + (lane & 15);
  const int kbB = (lane >> 4) << 3;
  const unsigned short* pa0 = sA + row * SAS + aCol + kbA;
  const unsigned short* pa1 = pa0 + 16 * SAS;
  const unsigned short* pb0 = W + (size_t)n * (size_t)ldk + kbB;
  acc0 = {}; acc1 = {};
  for (int kt = 0; kt < K; kt += 32) {
    union { v16bf v; uint4 q[2]; } ua0, ua1, ub;
    ua0.q[0] = *(const uint4*)(pa0 + kt);
    ua0.q[1] = *(const uint4*)(pa0 + kt + 16);
    ua1.q[0] = *(const uint4*)(pa1 + kt);
    ua1.q[1] = *(const uint4*)(pa1 + kt + 16);
    ub.q[0]  = *(const uint4*)(pb0 + kt);
    ub.q[1]  = *(const uint4*)(pb0 + kt + 16);
    acc0 = __builtin_amdgcn_wmma_f32_16x16x32_bf16(false, ua0.v, false, ub.v,
                                                   (short)0, acc0, false, false);
    acc1 = __builtin_amdgcn_wmma_f32_16x16x32_bf16(false, ua1.v, false, ub.v,
                                                   (short)0, acc1, false, false);
  }
}

// dynamic LDS layout
constexpr size_t SBUF_ELEMS   = (size_t)ROWS * SAS;          // bf16 activations
constexpr size_t SBUF_BYTES   = SBUF_ELEMS * 2;              // 82432
constexpr size_t SREF_ELEMS   = (size_t)ROWS * DMODEL;       // f32 residual
constexpr size_t SREF_BYTES   = SREF_ELEMS * 4;              // 32768
constexpr size_t SQK_BYTES    = DMODEL * 4;                  // 1024
constexpr size_t SSTATSP_ELEMS = 8 * ROWS * 2;               // per-wave partials
constexpr size_t SSTATSP_BYTES = SSTATSP_ELEMS * 4;          // 2048
constexpr size_t SSTATS_BYTES = ROWS * 2 * 4;                // 256
constexpr size_t SMEM_TOTAL = SBUF_BYTES + SREF_BYTES + SQK_BYTES +
                              SSTATSP_BYTES + SSTATS_BYTES;  // 118528 <= 320KB

extern __shared__ char smem_raw[] __attribute__((aligned(16)));

__global__ __launch_bounds__(256) void mrp_fused_kernel(Params P) {
  unsigned short* sBuf = (unsigned short*)smem_raw;
  float* sRef    = (float*)(smem_raw + SBUF_BYTES);
  float* sQk     = (float*)(smem_raw + SBUF_BYTES + SREF_BYTES);
  float* sStatsP = (float*)(smem_raw + SBUF_BYTES + SREF_BYTES + SQK_BYTES);
  float* sStats  = (float*)(smem_raw + SBUF_BYTES + SREF_BYTES + SQK_BYTES + SSTATSP_BYTES);

  const int tid  = threadIdx.x;
  const int wave = tid >> 5;
  const int lane = tid & 31;
  const int r0   = blockIdx.x * ROWS;
  const int rhalf = (lane >> 4) << 3; // C-fragment row base within strip (0|8)

  sQk[tid] = P.qk[tid];
  __syncthreads();

  for (int side = 0; side < 2; ++side) {
    const float* otherWalk = (side == 0) ? P.dst_walk : P.src_walk;
    const float* ownWalk   = (side == 0) ? P.src_walk : P.dst_walk;
    const float* pt        = (side == 0) ? P.src_pt   : P.dst_pt;
    const unsigned char* masks = (side == 0) ? P.src_masks : P.dst_masks;
    const unsigned short* Wv_c  = P.ws + P.wv_c[side];
    const unsigned short* Wo_c  = P.ws + P.wo_c[side];
    const unsigned short* Wg    = P.ws + P.wg[side];
    const unsigned short* Wv_p  = P.ws + P.wv_p;
    const unsigned short* Wo_p  = P.ws + P.wo_p;
    const unsigned short* Wf1   = P.ws + P.wf1[side];
    const unsigned short* Wf2   = P.ws + P.wf2[side];
    const unsigned short* Wffn1 = P.ws + P.wffn1[side];
    const unsigned short* Wffn2 = P.ws + P.wffn2[side];

    // ---- S0: other-side walk -> bf16 LDS cols [0,256) ----------------------
    for (int i = tid; i < ROWS * DMODEL; i += 256) {
      int row = i >> 8, col = i & 255;
      sBuf[row * SAS + col] = f32_to_bf16(otherWalk[(size_t)(r0 + row) * DMODEL + col]);
    }
    __syncthreads();

    // ---- S1: v-proj of cross: cols[0,256) -> cols[256,512) -----------------
    for (int nt = wave; nt < 16; nt += 8) {
      v8f a0, a1;
      wmma_tile2(sBuf, 0, 256, Wv_c, 256, nt * 16, lane, a0, a1);
      int n = nt * 16 + (lane & 15);
      float bb = P.bv_c[side][n];
#pragma unroll
      for (int r = 0; r < 8; ++r) {
        sBuf[(r + rhalf) * SAS + 256 + n]      = f32_to_bf16(a0[r] + bb);
        sBuf[(16 + r + rhalf) * SAS + 256 + n] = f32_to_bf16(a1[r] + bb);
      }
    }
    __syncthreads();

    // ---- S2: o-proj: cols[256,512) -> cross at cols[512,768) ---------------
    for (int nt = wave; nt < 16; nt += 8) {
      v8f a0, a1;
      wmma_tile2(sBuf, 256, 256, Wo_c, 256, nt * 16, lane, a0, a1);
      int n = nt * 16 + (lane & 15);
      float bb = P.bo_c[side][n];
#pragma unroll
      for (int r = 0; r < 8; ++r) {
        sBuf[(r + rhalf) * SAS + 512 + n]      = f32_to_bf16(a0[r] + bb);
        sBuf[(16 + r + rhalf) * SAS + 512 + n] = f32_to_bf16(a1[r] + bb);
      }
    }
    __syncthreads();
    // own walk -> cols [256,512): concat [walk, cross] spans cols [256,768)
    for (int i = tid; i < ROWS * DMODEL; i += 256) {
      int row = i >> 8, col = i & 255;
      sBuf[row * SAS + 256 + col] = f32_to_bf16(ownWalk[(size_t)(r0 + row) * DMODEL + col]);
    }
    __syncthreads();

    // ---- S3: gate GEMM K=512; refined = walk + sigmoid(g)*cross ------------
    for (int nt = wave; nt < 16; nt += 8) {
      v8f a[2];
      wmma_tile2(sBuf, 256, 512, Wg, 512, nt * 16, lane, a[0], a[1]);
      int n = nt * 16 + (lane & 15);
      float bb = P.bg[side][n];
#pragma unroll
      for (int s = 0; s < 2; ++s)
#pragma unroll
        for (int r = 0; r < 8; ++r) {
          int row = s * 16 + r + rhalf;
          float g  = 1.0f / (1.0f + __expf(-(a[s][r] + bb)));
          float cr = bf16_to_f32(sBuf[row * SAS + 512 + n]);
          float wv = ownWalk[(size_t)(r0 + row) * DMODEL + n];
          sRef[row * DMODEL + n] = wv + g * cr;
        }
    }
    __syncthreads();

    // ---- S4: per-type pooling + v/o projections ----------------------------
    for (int t = 0; t < TT; ++t) {
      { // 4a: softmax-pool raw kv -> bf16 cols [0,256); 4 rows per wave
        int row = wave * 4;
        for (int rr = 0; rr < 4; ++rr, ++row) {
          size_t b = (size_t)(r0 + row);
          const float* base = pt + (b * (TT * MM) + (size_t)t * MM) * DMODEL;
          const unsigned char* mk = masks + (b * TT + (size_t)t) * MM;
          int d0 = lane * 8;
          if (t + 1 < TT)
            __builtin_prefetch(base + (size_t)MM * DMODEL + d0, 0, 0);
          float kvv[MM][8]; float pm[MM];
#pragma unroll
          for (int m = 0; m < MM; ++m) {
            const float4* p = (const float4*)(base + m * DMODEL + d0);
            float4 x0 = p[0], x1 = p[1];
            kvv[m][0] = x0.x; kvv[m][1] = x0.y; kvv[m][2] = x0.z; kvv[m][3] = x0.w;
            kvv[m][4] = x1.x; kvv[m][5] = x1.y; kvv[m][6] = x1.z; kvv[m][7] = x1.w;
            float s = 0.f;
#pragma unroll
            for (int j = 0; j < 8; ++j) s += sQk[d0 + j] * kvv[m][j];
            pm[m] = s;
          }
#pragma unroll
          for (int m = 0; m < MM; ++m)
            for (int off = 16; off > 0; off >>= 1)
              pm[m] += __shfl_xor(pm[m], off, 32);
          float mx = -3.4e38f;
#pragma unroll
          for (int m = 0; m < MM; ++m)
            if (mk[m] && pm[m] > mx) mx = pm[m];
          float w[MM]; float den = 0.f;
#pragma unroll
          for (int m = 0; m < MM; ++m) {
            w[m] = mk[m] ? __expf(pm[m] - mx) : 0.f;
            den += w[m];
          }
          float inv = den > 0.f ? 1.f / den : 0.f;
#pragma unroll
          for (int j = 0; j < 8; ++j) {
            float pv = 0.f;
#pragma unroll
            for (int m = 0; m < MM; ++m) pv += w[m] * kvv[m][j];
            sBuf[row * SAS + d0 + j] = f32_to_bf16(pv * inv);
          }
        }
      }
      __syncthreads();
      // 4b: v-proj pooled kv -> cols [1024,1280)
      for (int nt = wave; nt < 16; nt += 8) {
        v8f a0, a1;
        wmma_tile2(sBuf, 0, 256, Wv_p, 256, nt * 16, lane, a0, a1);
        int n = nt * 16 + (lane & 15);
        float bb = P.bv_p[n];
#pragma unroll
        for (int r = 0; r < 8; ++r) {
          sBuf[(r + rhalf) * SAS + 1024 + n]      = f32_to_bf16(a0[r] + bb);
          sBuf[(16 + r + rhalf) * SAS + 1024 + n] = f32_to_bf16(a1[r] + bb);
        }
      }
      __syncthreads();
      // 4c: o-proj -> flat block t at cols [256+256t, 512+256t)
      for (int nt = wave; nt < 16; nt += 8) {
        v8f a0, a1;
        wmma_tile2(sBuf, 1024, 256, Wo_p, 256, nt * 16, lane, a0, a1);
        int n = nt * 16 + (lane & 15);
        float bb = P.bo_p[n];
#pragma unroll
        for (int r = 0; r < 8; ++r) {
          sBuf[(r + rhalf) * SAS + 256 + t * 256 + n]      = f32_to_bf16(a0[r] + bb);
          sBuf[(16 + r + rhalf) * SAS + 256 + t * 256 + n] = f32_to_bf16(a1[r] + bb);
        }
      }
      __syncthreads();
    }

    // ---- S5: fused-1 K=768 (cols[256,1024)) -> LN -> ReLU -> cols[0,512) ---
    {
      v8f hacc[4][2];
      float ls[2][8], lq[2][8];
#pragma unroll
      for (int s = 0; s < 2; ++s)
#pragma unroll
        for (int r = 0; r < 8; ++r) { ls[s][r] = 0.f; lq[s][r] = 0.f; }
      for (int i = 0; i < 4; ++i) {
        int nt = wave + 8 * i;
        wmma_tile2(sBuf, 256, 768, Wf1, 768, nt * 16, lane, hacc[i][0], hacc[i][1]);
        int n = nt * 16 + (lane & 15);
        float bb = P.bf1[side][n];
#pragma unroll
        for (int s = 0; s < 2; ++s)
#pragma unroll
          for (int r = 0; r < 8; ++r) {
            float v = hacc[i][s][r] + bb;
            hacc[i][s][r] = v;
            ls[s][r] += v; lq[s][r] += v * v;
          }
      }
#pragma unroll
      for (int off = 8; off >= 1; off >>= 1)
#pragma unroll
        for (int s = 0; s < 2; ++s)
#pragma unroll
          for (int r = 0; r < 8; ++r) {
            ls[s][r] += __shfl_xor(ls[s][r], off, 32);
            lq[s][r] += __shfl_xor(lq[s][r], off, 32);
          }
      if ((lane & 15) == 0)
#pragma unroll
        for (int s = 0; s < 2; ++s)
#pragma unroll
          for (int r = 0; r < 8; ++r) {
            int row = s * 16 + rhalf + r;
            sStatsP[(wave * ROWS + row) * 2 + 0] = ls[s][r];
            sStatsP[(wave * ROWS + row) * 2 + 1] = lq[s][r];
          }
      __syncthreads();
      if (tid < 64) {
        int row = tid >> 1, c = tid & 1;
        float s = 0.f;
        for (int w2 = 0; w2 < 8; ++w2) s += sStatsP[(w2 * ROWS + row) * 2 + c];
        sStats[row * 2 + c] = s;
      }
      __syncthreads();
      for (int i = 0; i < 4; ++i) {
        int nt = wave + 8 * i;
        int n = nt * 16 + (lane & 15);
        float g1 = P.lng1[side][n], b1 = P.lnb1[side][n];
#pragma unroll
        for (int s = 0; s < 2; ++s)
#pragma unroll
          for (int r = 0; r < 8; ++r) {
            int row = s * 16 + r + rhalf;
            float mean = sStats[row * 2 + 0] * (1.f / 512.f);
            float var  = sStats[row * 2 + 1] * (1.f / 512.f) - mean * mean;
            float v = (hacc[i][s][r] - mean) * rsqrtf(var + 1e-5f) * g1 + b1;
            v = v > 0.f ? v : 0.f;
            sBuf[row * SAS + n] = f32_to_bf16(v);
          }
      }
    }
    __syncthreads();

    // ---- S6: fused-2 K=512 -> refined += out -------------------------------
    for (int nt = wave; nt < 16; nt += 8) {
      v8f a[2];
      wmma_tile2(sBuf, 0, 512, Wf2, 512, nt * 16, lane, a[0], a[1]);
      int n = nt * 16 + (lane & 15);
      float bb = P.bf2[side][n];
#pragma unroll
      for (int s = 0; s < 2; ++s)
#pragma unroll
        for (int r = 0; r < 8; ++r) {
          int row = s * 16 + r + rhalf;
          sRef[row * DMODEL + n] += a[s][r] + bb;
        }
    }
    __syncthreads();

    // ---- S7: refined -> bf16 cols[0,256) -----------------------------------
    for (int i = tid; i < ROWS * DMODEL; i += 256)
      sBuf[(i >> 8) * SAS + (i & 255)] = f32_to_bf16(sRef[i]);
    __syncthreads();

    // ---- S8: FFN-1 K=256 N=1024 -> exact GELU -> cols[256,1280) ------------
    for (int i = 0; i < 8; ++i) {
      int nt = wave + 8 * i;
      v8f a[2];
      wmma_tile2(sBuf, 0, 256, Wffn1, 256, nt * 16, lane, a[0], a[1]);
      int n = nt * 16 + (lane & 15);
      float bb = P.bffn1[side][n];
#pragma unroll
      for (int s = 0; s < 2; ++s)
#pragma unroll
        for (int r = 0; r < 8; ++r) {
          float x = a[s][r] + bb;
          float g = 0.5f * x * (1.0f + erff(x * 0.70710678118654752f));
          sBuf[(s * 16 + r + rhalf) * SAS + 256 + n] = f32_to_bf16(g);
        }
    }
    __syncthreads();

    // ---- S9: FFN-2 K=1024 + residual -> LN -> clip -> global ---------------
    {
      v8f facc[2][2];
      float ls[2][8], lq[2][8];
#pragma unroll
      for (int s = 0; s < 2; ++s)
#pragma unroll
        for (int r = 0; r < 8; ++r) { ls[s][r] = 0.f; lq[s][r] = 0.f; }
      for (int i = 0; i < 2; ++i) {
        int nt = wave + 8 * i;
        wmma_tile2(sBuf, 256, 1024, Wffn2, 1024, nt * 16, lane, facc[i][0], facc[i][1]);
        int n = nt * 16 + (lane & 15);
        float bb = P.bffn2[side][n];
#pragma unroll
        for (int s = 0; s < 2; ++s)
#pragma unroll
          for (int r = 0; r < 8; ++r) {
            int row = s * 16 + r + rhalf;
            float v = sRef[row * DMODEL + n] + facc[i][s][r] + bb;
            facc[i][s][r] = v;
            ls[s][r] += v; lq[s][r] += v * v;
          }
      }
#pragma unroll
      for (int off = 8; off >= 1; off >>= 1)
#pragma unroll
        for (int s = 0; s < 2; ++s)
#pragma unroll
          for (int r = 0; r < 8; ++r) {
            ls[s][r] += __shfl_xor(ls[s][r], off, 32);
            lq[s][r] += __shfl_xor(lq[s][r], off, 32);
          }
      if ((lane & 15) == 0)
#pragma unroll
        for (int s = 0; s < 2; ++s)
#pragma unroll
          for (int r = 0; r < 8; ++r) {
            int row = s * 16 + rhalf + r;
            sStatsP[(wave * ROWS + row) * 2 + 0] = ls[s][r];
            sStatsP[(wave * ROWS + row) * 2 + 1] = lq[s][r];
          }
      __syncthreads();
      if (tid < 64) {
        int row = tid >> 1, c = tid & 1;
        float s = 0.f;
        for (int w2 = 0; w2 < 8; ++w2) s += sStatsP[(w2 * ROWS + row) * 2 + c];
        sStats[row * 2 + c] = s;
      }
      __syncthreads();
      float* outp = P.out + ((size_t)side * BB + (size_t)r0) * DMODEL;
      for (int i = 0; i < 2; ++i) {
        int nt = wave + 8 * i;
        int n = nt * 16 + (lane & 15);
        float g = P.lng[side][n], bb = P.lnb[side][n];
#pragma unroll
        for (int s = 0; s < 2; ++s)
#pragma unroll
          for (int r = 0; r < 8; ++r) {
            int row = s * 16 + r + rhalf;
            float mean = sStats[row * 2 + 0] * (1.f / 256.f);
            float var  = sStats[row * 2 + 1] * (1.f / 256.f) - mean * mean;
            float y = (facc[i][s][r] - mean) * rsqrtf(var + 1e-5f) * g + bb;
            if (isnan(y)) y = 0.f;
            else if (isinf(y)) y = y > 0.f ? 10.f : -10.f;
            outp[(size_t)row * DMODEL + n] = y;
          }
      }
    }
    __syncthreads();
  }
}

// ---- prep kernels ----------------------------------------------------------
__global__ void cvt_bf16_kernel(const float* __restrict__ src,
                                unsigned short* __restrict__ dst, int n) {
  int i = blockIdx.x * 256 + threadIdx.x;
  if (i < n) dst[i] = f32_to_bf16(src[i]);
}

// qk[e] = (1/sqrt(D)) * sum_d (pq @ Wq_p^T + bq_p)[d] * Wk_p[d][e]
__global__ void qk_kernel(const float* __restrict__ pq, const float* __restrict__ Wq,
                          const float* __restrict__ bq, const float* __restrict__ Wk,
                          float* __restrict__ qk) {
  __shared__ float q[DMODEL];
  int d = threadIdx.x;
  float s = bq[d];
  for (int e = 0; e < DMODEL; ++e) s += pq[e] * Wq[d * DMODEL + e];
  q[d] = s;
  __syncthreads();
  float s2 = 0.f;
  for (int dd = 0; dd < DMODEL; ++dd) s2 += q[dd] * Wk[dd * DMODEL + d];
  qk[d] = s2 * (1.0f / 16.0f);
}

// ---- workspace layout (bf16 element offsets) -------------------------------
enum : size_t {
  OFF_WV_S2D = 0,        OFF_WO_S2D = 65536,
  OFF_WV_D2S = 131072,   OFF_WO_D2S = 196608,
  OFF_WG_S   = 262144,   OFF_WG_D   = 393216,
  OFF_WV_P   = 524288,   OFF_WO_P   = 589824,
  OFF_WF1_S  = 655360,   OFF_WF2_S  = 1048576,
  OFF_WF1_D  = 1179648,  OFF_WF2_D  = 1572864,
  OFF_WFFN1_S = 1703936, OFF_WFFN2_S = 1966080,
  OFF_WFFN1_D = 2228224, OFF_WFFN2_D = 2490368,
  TOTAL_WELEMS = 2752512
};

extern "C" void kernel_launch(void* const* d_in, const int* in_sizes, int n_in,
                              void* d_out, int out_size, void* d_ws, size_t ws_size,
                              hipStream_t stream) {
  auto F = [&](int i) { return (const float*)d_in[i]; };
  unsigned short* ws = (unsigned short*)d_ws;
  float* qk = (float*)((char*)d_ws + (size_t)TOTAL_WELEMS * 2);

  struct Job { int idx; size_t off; int n; };
  const Job jobs[] = {
    {4,  OFF_WV_S2D, 65536},  {6,  OFF_WO_S2D, 65536},
    {8,  OFF_WV_D2S, 65536},  {10, OFF_WO_D2S, 65536},
    {12, OFF_WG_S, 131072},   {14, OFF_WG_D, 131072},
    {21, OFF_WV_P, 65536},    {23, OFF_WO_P, 65536},
    {25, OFF_WF1_S, 393216},  {29, OFF_WF2_S, 131072},
    {31, OFF_WF1_D, 393216},  {35, OFF_WF2_D, 131072},
    {37, OFF_WFFN1_S, 262144},{39, OFF_WFFN2_S, 262144},
    {41, OFF_WFFN1_D, 262144},{43, OFF_WFFN2_D, 262144},
  };
  for (const Job& j : jobs)
    cvt_bf16_kernel<<<(j.n + 255) / 256, 256, 0, stream>>>(F(j.idx), ws + j.off, j.n);

  qk_kernel<<<1, 256, 0, stream>>>(F(16), F(17), F(18), F(19), qk);

  Params P;
  P.out = (float*)d_out;
  P.src_walk = F(0); P.dst_walk = F(1);
  P.src_pt = F(2);   P.dst_pt = F(3);
  P.src_masks = (const unsigned char*)d_in[49];
  P.dst_masks = (const unsigned char*)d_in[50];
  P.ws = ws; P.qk = qk;
  P.bv_c[0] = F(5);  P.bv_c[1] = F(9);
  P.bo_c[0] = F(7);  P.bo_c[1] = F(11);
  P.bg[0]  = F(13);  P.bg[1]  = F(15);
  P.bv_p = F(22);    P.bo_p = F(24);
  P.bf1[0] = F(26);  P.bf1[1] = F(32);
  P.lng1[0] = F(27); P.lng1[1] = F(33);
  P.lnb1[0] = F(28); P.lnb1[1] = F(34);
  P.bf2[0] = F(30);  P.bf2[1] = F(36);
  P.bffn1[0] = F(38); P.bffn1[1] = F(42);
  P.bffn2[0] = F(40); P.bffn2[1] = F(44);
  P.lng[0] = F(45);  P.lng[1] = F(47);
  P.lnb[0] = F(46);  P.lnb[1] = F(48);
  P.wv_c[0] = OFF_WV_S2D; P.wv_c[1] = OFF_WV_D2S;
  P.wo_c[0] = OFF_WO_S2D; P.wo_c[1] = OFF_WO_D2S;
  P.wg[0] = OFF_WG_S;     P.wg[1] = OFF_WG_D;
  P.wf1[0] = OFF_WF1_S;   P.wf1[1] = OFF_WF1_D;
  P.wf2[0] = OFF_WF2_S;   P.wf2[1] = OFF_WF2_D;
  P.wffn1[0] = OFF_WFFN1_S; P.wffn1[1] = OFF_WFFN1_D;
  P.wffn2[0] = OFF_WFFN2_S; P.wffn2[1] = OFF_WFFN2_D;
  P.wv_p = OFF_WV_P;      P.wo_p = OFF_WO_P;

  mrp_fused_kernel<<<BB / ROWS, 256, (unsigned)SMEM_TOTAL, stream>>>(P);
}
